// HANLayer_18176301597371
// MI455X (gfx1250) — matrix operand order repaired and checked by reading, not compiled, not run on hardware
//
#include <hip/hip_runtime.h>

#define NN   30000
#define EE   480000
#define PP   3
#define IND  512
#define HID  128
#define OUTD 128
#define NEG_SLOPE 0.2f

typedef __attribute__((ext_vector_type(16))) _Float16 v16h;
typedef __attribute__((ext_vector_type(8)))  _Float16 v8h;
typedef __attribute__((ext_vector_type(8)))  float    v8f;

#define LDS_STRIDE 40   // halves per row: 80B => conflict-free & 16B-aligned fragments

// ---------------- utility ----------------
__device__ inline void atomicMaxF(float* addr, float val) {
    unsigned int* ua = (unsigned int*)addr;
    unsigned int old = *ua;
    while (true) {
        float oldf = __uint_as_float(old);
        if (oldf >= val) break;
        unsigned int assumed = old;
        old = atomicCAS(ua, assumed, __float_as_uint(val));
        if (old == assumed) break;
    }
}

__global__ void fill_kernel(float* p, float v, int n) {
    int i = blockIdx.x * blockDim.x + threadIdx.x;
    if (i < n) p[i] = v;
}

// ---------------- WMMA GEMM: C[M,128] = A[M,K] @ B[K,128] (f32 in/out, f16 compute) ----
// Block: 256 threads (8 waves) -> 128x128 C tile. Wave w -> rows w*16..w*16+15, all 8 N-tiles.
// LDS-staged A (128x32) and B (32x128) per K-step; 8 WMMAs per wave per K-step.
// epi: 0 = plain store, 1 = relu(x + bias[col])
__global__ __launch_bounds__(256) void gemm_wmma_kernel(
    const float* __restrict__ A, const float* __restrict__ B, float* __restrict__ C,
    int M, int K, int lda, int ldb, int ldc, int epi, const float* __restrict__ bias)
{
    __shared__ _Float16 As[128 * LDS_STRIDE];   // [row 0..127][K 0..31]
    __shared__ _Float16 Bs[128 * LDS_STRIDE];   // [col 0..127][K 0..31]  (transposed)

    const int t       = threadIdx.x;
    const int w       = t >> 5;
    const int lane    = t & 31;
    const int row16   = lane & 15;
    const int half    = lane >> 4;
    const int m_block = blockIdx.x * 128;

    v8f acc[8];
#pragma unroll
    for (int i = 0; i < 8; ++i) acc[i] = (v8f){};

    for (int k0 = 0; k0 < K; k0 += 32) {
        // ---- stage A tile: 128 rows x 32 K (1024 float4 loads, coalesced) ----
#pragma unroll
        for (int i = 0; i < 4; ++i) {
            int v   = t + i * 256;              // 0..1023
            int ar  = v >> 3;                   // row in tile 0..127
            int ak4 = (v & 7) * 4;              // K offset 0..28
            int gr  = m_block + ar; if (gr > M - 1) gr = M - 1;   // clamp OOB rows
            float4 x = *(const float4*)(A + (size_t)gr * lda + k0 + ak4);
            _Float16* dstp = As + ar * LDS_STRIDE + ak4;
            dstp[0] = (_Float16)x.x; dstp[1] = (_Float16)x.y;
            dstp[2] = (_Float16)x.z; dstp[3] = (_Float16)x.w;
        }
        // ---- stage B tile transposed: 32 K x 128 cols ----
#pragma unroll
        for (int i = 0; i < 4; ++i) {
            int v   = t + i * 256;              // 0..1023
            int kr  = v >> 5;                   // K row 0..31
            int c4  = (v & 31) * 4;             // col 0..124
            float4 x = *(const float4*)(B + (size_t)(k0 + kr) * ldb + c4);
            Bs[(c4 + 0) * LDS_STRIDE + kr] = (_Float16)x.x;
            Bs[(c4 + 1) * LDS_STRIDE + kr] = (_Float16)x.y;
            Bs[(c4 + 2) * LDS_STRIDE + kr] = (_Float16)x.z;
            Bs[(c4 + 3) * LDS_STRIDE + kr] = (_Float16)x.w;
        }
        __syncthreads();

        // ---- A fragment (ISA layout): lane holds row=lane%16, K = half*8+{0..7}, 16+half*8+{0..7}
        const _Float16* arow = As + (w * 16 + row16) * LDS_STRIDE;
        v8h alo = *(const v8h*)(arow + half * 8);
        v8h ahi = *(const v8h*)(arow + 16 + half * 8);
        v16h av = __builtin_shufflevector(alo, ahi, 0,1,2,3,4,5,6,7,8,9,10,11,12,13,14,15);

        // ---- 8 N-tiles: B fragment lane holds col=tile*16+lane%16, K = half*8+{0..7}, 16+half*8+{0..7}
#pragma unroll
        for (int nt = 0; nt < 8; ++nt) {
            const _Float16* bcol = Bs + (nt * 16 + row16) * LDS_STRIDE;
            v8h blo = *(const v8h*)(bcol + half * 8);
            v8h bhi = *(const v8h*)(bcol + 16 + half * 8);
            v16h bv = __builtin_shufflevector(blo, bhi, 0,1,2,3,4,5,6,7,8,9,10,11,12,13,14,15);
            acc[nt] = __builtin_amdgcn_wmma_f32_16x16x32_f16(
                false, av, false, bv, (short)0, acc[nt], false, false);
        }
        __syncthreads();
    }

    // ---- epilogue: C row = m_block + w*16 + half*8 + j, col = nt*16 + lane%16 ----
#pragma unroll
    for (int nt = 0; nt < 8; ++nt) {
        int c = nt * 16 + row16;
        float bcol = (epi == 1) ? bias[c] : 0.0f;
#pragma unroll
        for (int j = 0; j < 8; ++j) {
            int r = m_block + w * 16 + half * 8 + j;
            if (r < M) {
                float v = acc[nt][j];
                if (epi == 1) { v += bcol; v = v > 0.0f ? v : 0.0f; }
                C[(size_t)r * ldc + c] = v;
            }
        }
    }
}

// ---------------- per-node attention scalars: el = f.al, er = f.ar ----------------
__global__ void attn_scores_kernel(const float* __restrict__ f,
                                   const float* __restrict__ al, const float* __restrict__ ar,
                                   float* __restrict__ el, float* __restrict__ er, int n)
{
    int i = blockIdx.x * blockDim.x + threadIdx.x;
    if (i >= n) return;
    const float4* fr = (const float4*)(f + (size_t)i * HID);
    const float4* av = (const float4*)al;
    const float4* bv = (const float4*)ar;
    float sl = 0.f, sr = 0.f;
#pragma unroll 8
    for (int k = 0; k < HID / 4; ++k) {
        float4 fv = fr[k], a = av[k], b = bv[k];
        sl += fv.x * a.x + fv.y * a.y + fv.z * a.z + fv.w * a.w;
        sr += fv.x * b.x + fv.y * b.y + fv.z * b.z + fv.w * b.w;
    }
    el[i] = sl; er[i] = sr;
}

// ---------------- edge softmax pass 1: e = leaky_relu(el[src]+er[dst]); m[dst] = max ----
__global__ void edge_max_kernel(const int* __restrict__ src, const int* __restrict__ dst,
                                const float* __restrict__ el, const float* __restrict__ er,
                                float* __restrict__ ebuf, float* __restrict__ m, int E)
{
    int i = blockIdx.x * blockDim.x + threadIdx.x;
    if (i >= E) return;
    float x = el[src[i]] + er[dst[i]];
    float e = x > 0.0f ? x : NEG_SLOPE * x;
    ebuf[i] = e;
    atomicMaxF(&m[dst[i]], e);
}

// ---------------- edge softmax pass 2: a = exp(e - m[dst]); s[dst] += a ----------------
__global__ void edge_expsum_kernel(const float* __restrict__ ebuf, const int* __restrict__ dst,
                                   const float* __restrict__ m, float* __restrict__ s,
                                   float* __restrict__ abuf, int E)
{
    int i = blockIdx.x * blockDim.x + threadIdx.x;
    if (i >= E) return;
    int d = dst[i];
    float a = __expf(ebuf[i] - m[d]);
    abuf[i] = a;
    atomicAdd(&s[d], a);
}

// ---------------- edge aggregation: acc[dst] += f[src] * (a / s[dst]) ----------------
__global__ void edge_agg_kernel(const int* __restrict__ src, const int* __restrict__ dst,
                                const float* __restrict__ abuf, const float* __restrict__ s,
                                const float* __restrict__ f, float* __restrict__ acc, int E)
{
    int gid = blockIdx.x * blockDim.x + threadIdx.x;
    int e = gid >> 5;
    if (e >= E) return;
    int lane = gid & 31;
    int sn = src[e], dn = dst[e];
    float w = abuf[e] / s[dn];
    int k = lane * 4;
    float4 fv = *(const float4*)(f + (size_t)sn * HID + k);
    float* ap = acc + (size_t)dn * HID + k;
    atomicAdd(ap + 0, fv.x * w);
    atomicAdd(ap + 1, fv.y * w);
    atomicAdd(ap + 2, fv.z * w);
    atomicAdd(ap + 3, fv.w * w);
}

// ---------------- node epilogue: out = tanh(acc + b), strided store ----------------
__global__ void node_finish_kernel(const float* __restrict__ acc, const float* __restrict__ bias,
                                   float* __restrict__ out, int n, int ostride, int ooff)
{
    int i = blockIdx.x * blockDim.x + threadIdx.x;
    if (i >= n * HID) return;
    int node = i >> 7, k = i & 127;
    out[(size_t)node * ostride + ooff + k] = tanhf(acc[i] + bias[k]);
}

// ---------------- semantic score: sc[row] = dot(T[row], Ws2) + bs2 ----------------
__global__ void sem_score_kernel(const float* __restrict__ T, const float* __restrict__ Ws2,
                                 const float* __restrict__ bs2, float* __restrict__ sc, int rows)
{
    int i = blockIdx.x * blockDim.x + threadIdx.x;
    if (i >= rows) return;
    const float4* tr = (const float4*)(T + (size_t)i * HID);
    const float4* wv = (const float4*)Ws2;
    float acc = 0.f;
#pragma unroll 8
    for (int k = 0; k < HID / 4; ++k) {
        float4 tv = tr[k], w = wv[k];
        acc += tv.x * w.x + tv.y * w.y + tv.z * w.z + tv.w * w.w;
    }
    sc[i] = acc + bs2[0];
}

// ---------------- semantic softmax over P and weighted output ----------------
__global__ void sem_out_kernel(const float* __restrict__ Z, const float* __restrict__ sc,
                               float* __restrict__ out, int n)
{
    int i = blockIdx.x * blockDim.x + threadIdx.x;
    if (i >= n * HID) return;
    int node = i >> 7, k = i & 127;
    float s0 = sc[node * PP + 0], s1 = sc[node * PP + 1], s2 = sc[node * PP + 2];
    float mx = fmaxf(s0, fmaxf(s1, s2));
    float e0 = __expf(s0 - mx), e1 = __expf(s1 - mx), e2 = __expf(s2 - mx);
    float inv = 1.0f / (e0 + e1 + e2);
    size_t base = (size_t)node * (PP * HID) + k;
    out[base + 0 * HID] = e0 * inv * Z[base + 0 * HID];
    out[base + 1 * HID] = e1 * inv * Z[base + 1 * HID];
    out[base + 2 * HID] = e2 * inv * Z[base + 2 * HID];
}

// ====================================================================================
extern "C" void kernel_launch(void* const* d_in, const int* in_sizes, int n_in,
                              void* d_out, int out_size, void* d_ws, size_t ws_size,
                              hipStream_t stream)
{
    const float* h   = (const float*)d_in[0];
    const int*   src = (const int*)  d_in[1];
    const int*   dst = (const int*)  d_in[2];
    const float* W1  = (const float*)d_in[3];
    const float* al1 = (const float*)d_in[4];
    const float* ar1 = (const float*)d_in[5];
    const float* b1  = (const float*)d_in[6];
    const float* W2  = (const float*)d_in[7];
    const float* al2 = (const float*)d_in[8];
    const float* ar2 = (const float*)d_in[9];
    const float* b2  = (const float*)d_in[10];
    const float* Ws1 = (const float*)d_in[11];
    const float* bs1 = (const float*)d_in[12];
    const float* Ws2 = (const float*)d_in[13];
    const float* bs2 = (const float*)d_in[14];
    float* out = (float*)d_out;

    // ---- workspace layout (floats) ----
    float* ws = (float*)d_ws;
    size_t off = 0;
    float* Z    = ws + off; off += (size_t)NN * PP * HID;   // 11.52M
    float* fbuf = ws + off; off += (size_t)NN * HID;        //  3.84M
    float* z1   = ws + off; off += (size_t)NN * HID;        //  3.84M
    float* acc  = ws + off; off += (size_t)NN * HID;        //  3.84M
    float* Tbuf = fbuf;                                     // reused after metapath loop
    float* m    = ws + off; off += NN;
    float* svec = ws + off; off += NN;
    float* el   = ws + off; off += NN;
    float* er   = ws + off; off += NN;
    float* ebuf = ws + off; off += EE;
    float* abuf = ws + off; off += EE;
    float* sc   = ws + off; off += (size_t)NN * PP;

    const int B256 = 256;
    auto blk  = [](long n) { return (int)((n + 255) / 256); };
    auto gblk = [](long m) { return (int)((m + 127) / 128); };   // 128-row GEMM blocks

    for (int p = 0; p < PP; ++p) {
        const int* sp = src + (size_t)p * EE;
        const int* dp = dst + (size_t)p * EE;

        // ===== GAT layer 1: f = h @ W1[p]  (M=30000, K=512, N=128) =====
        gemm_wmma_kernel<<<gblk(NN), 256, 0, stream>>>(
            h, W1 + (size_t)p * IND * HID, fbuf, NN, IND, IND, HID, HID, 0, nullptr);
        attn_scores_kernel<<<blk(NN), B256, 0, stream>>>(fbuf, al1 + p * HID, ar1 + p * HID, el, er, NN);
        fill_kernel<<<blk(NN), B256, 0, stream>>>(m, -3.402823466e38f, NN);
        fill_kernel<<<blk(NN), B256, 0, stream>>>(svec, 0.0f, NN);
        fill_kernel<<<blk((long)NN * HID), B256, 0, stream>>>(acc, 0.0f, NN * HID);
        edge_max_kernel<<<blk(EE), B256, 0, stream>>>(sp, dp, el, er, ebuf, m, EE);
        edge_expsum_kernel<<<blk(EE), B256, 0, stream>>>(ebuf, dp, m, svec, abuf, EE);
        edge_agg_kernel<<<blk((long)EE * 32), B256, 0, stream>>>(sp, dp, abuf, svec, fbuf, acc, EE);
        node_finish_kernel<<<blk((long)NN * HID), B256, 0, stream>>>(acc, b1 + p * HID, z1, NN, HID, 0);

        // ===== GAT layer 2: f = z1 @ W2[p]  (M=30000, K=128, N=128) =====
        gemm_wmma_kernel<<<gblk(NN), 256, 0, stream>>>(
            z1, W2 + (size_t)p * HID * OUTD, fbuf, NN, HID, HID, OUTD, OUTD, 0, nullptr);
        attn_scores_kernel<<<blk(NN), B256, 0, stream>>>(fbuf, al2 + p * OUTD, ar2 + p * OUTD, el, er, NN);
        fill_kernel<<<blk(NN), B256, 0, stream>>>(m, -3.402823466e38f, NN);
        fill_kernel<<<blk(NN), B256, 0, stream>>>(svec, 0.0f, NN);
        fill_kernel<<<blk((long)NN * HID), B256, 0, stream>>>(acc, 0.0f, NN * HID);
        edge_max_kernel<<<blk(EE), B256, 0, stream>>>(sp, dp, el, er, ebuf, m, EE);
        edge_expsum_kernel<<<blk(EE), B256, 0, stream>>>(ebuf, dp, m, svec, abuf, EE);
        edge_agg_kernel<<<blk((long)EE * 32), B256, 0, stream>>>(sp, dp, abuf, svec, fbuf, acc, EE);
        // write z2 directly into Z[:, p, :]  (row stride P*HID, col offset p*HID)
        node_finish_kernel<<<blk((long)NN * HID), B256, 0, stream>>>(acc, b2 + p * OUTD, Z, NN, PP * HID, p * HID);
    }

    // ===== semantic attention =====
    // T = relu(Z(view [N*P,128]) @ Ws1 + bs1)   M=90000, K=128, N=128
    gemm_wmma_kernel<<<gblk((long)NN * PP), 256, 0, stream>>>(
        Z, Ws1, Tbuf, NN * PP, HID, HID, HID, HID, 1, bs1);
    sem_score_kernel<<<blk((long)NN * PP), B256, 0, stream>>>(Tbuf, Ws2, bs2, sc, NN * PP);
    sem_out_kernel<<<blk((long)NN * HID), B256, 0, stream>>>(Z, sc, out, NN);
}
